// DSQGAttentionN_Fused_25451976196241
// MI455X (gfx1250) — matrix operand order repaired and checked by reading, not compiled, not run on hardware
//
#include <hip/hip_runtime.h>
#include <hip/hip_bf16.h>

typedef __attribute__((ext_vector_type(16))) _Float16 v16h;
typedef __attribute__((ext_vector_type(8)))  float    v8f;
typedef __attribute__((ext_vector_type(4)))  float    v4f;

#define NQ    4096
#define HDIM  64
#define NHEAD 16
#define NOFF  44

// 33 dense causal offsets 0..32, then 11 sparse ones.
static __device__ const int OFFS[NOFF] = {
    0, 1, 2, 3, 4, 5, 6, 7, 8, 9, 10, 11, 12, 13, 14, 15, 16,
    17, 18, 19, 20, 21, 22, 23, 24, 25, 26, 27, 28, 29, 30, 31, 32,
    48, 64, 96, 128, 192, 256, 384, 512, 768, 1024, 1536};

// Convert 8 consecutive scaled floats into halves a[h0..h0+7].
__device__ __forceinline__ void cvt8s(const float* p, float s, v16h& a, int h0) {
  const v4f* q = (const v4f*)p;
  v4f u = q[0], w = q[1];
#pragma unroll
  for (int j = 0; j < 4; ++j) {
    a[h0 + j]     = (_Float16)(u[j] * s);
    a[h0 + 4 + j] = (_Float16)(w[j] * s);
  }
}

// Unscaled variant (pure v_cvt_pk_f16_f32 path) for K fragments.
__device__ __forceinline__ void cvt8(const float* p, v16h& a, int h0) {
  const v4f* q = (const v4f*)p;
  v4f u = q[0], w = q[1];
#pragma unroll
  for (int j = 0; j < 4; ++j) {
    a[h0 + j]     = (_Float16)u[j];
    a[h0 + 4 + j] = (_Float16)w[j];
  }
}

__global__ __launch_bounds__(256)
void dsqg_attn_wmma_kernel(const float* __restrict__ q, const float* __restrict__ k,
                           const float* __restrict__ v, const float* __restrict__ bias,
                           float* __restrict__ out) {
  // Per-wave score table: [16 queries][44 offsets], padded to 64 so slots 48..63
  // act as a write "dump" for branchless predicated scatter. Init -inf.
  __shared__ float scw[8][16][64];

  const int lane = threadIdx.x & 31;
  const int w    = threadIdx.x >> 5;
  const int bid  = blockIdx.x;
  const int bh   = bid >> 5;           // 32 blocks of 128 queries per (b,h)
  const int n0   = (bid & 31) * 128;
  const int hh   = bh & (NHEAD - 1);
  const int t    = n0 + w * 16;        // this wave's 16-query tile base

  const size_t base = (size_t)bh * NQ * HDIM;
  const float* qb = q + base;
  const float* kb = k + base;
  const float* vb = v + base;

  const int Nl = lane & 15;
  const int hi = lane >> 4;

  {  // init this wave's score region to -inf (causal mask falls out of softmax)
    float* p = &scw[w][0][0];
    for (int idx = lane; idx < 16 * 64; idx += 32) p[idx] = -__builtin_inff();
  }

  // ---- A fragments: Q tile (16x64), scale 1/sqrt(64)=0.125 folded into f16 cvt.
  // A layout: lane l -> row M=l&15; halves cover K = 16*hi..16*hi+15 (+32 for a1).
  const float* qrow = qb + (size_t)(t + Nl) * HDIM;
  v16h a0, a1;
  cvt8s(qrow +      16 * hi,     0.125f, a0, 0);
  cvt8s(qrow +      16 * hi + 8, 0.125f, a0, 8);
  cvt8s(qrow + 32 + 16 * hi,     0.125f, a1, 0);
  cvt8s(qrow + 32 + 16 * hi + 8, 0.125f, a1, 8);

  // ---- Dense offsets d=0..32: banded 16x48 = Q(16x64) * K[t-32..t+15]^T via 3x2 WMMA.
#pragma unroll
  for (int blk = 0; blk < 3; ++blk) {
    const int baserow = t - 32 + 16 * blk;
    const int krow    = baserow + Nl;          // B column N=l&15 holds k-row baserow+N
    v16h b0 = {}, b1 = {};
    if (krow >= 0) {
      const float* kr = kb + (size_t)krow * HDIM;
      cvt8(kr +      16 * hi,     b0, 0);      // B layout: K = h + 16*hi
      cvt8(kr +      16 * hi + 8, b0, 8);
      cvt8(kr + 32 + 16 * hi,     b1, 0);
      cvt8(kr + 32 + 16 * hi + 8, b1, 8);
      __builtin_prefetch(vb + (size_t)krow * HDIM, 0, 1);  // warm v rows for phase 2
    }
    v8f c = {};
    c = __builtin_amdgcn_wmma_f32_16x16x32_f16(false, a0, false, b0, (short)0, c, false, false);
    c = __builtin_amdgcn_wmma_f32_16x16x32_f16(false, a1, false, b1, (short)0, c, false, false);
    // D layout: lane l has N=l&15, VGPR vv has M=vv+8*hi.  d = M + 32 - jg.
    // Branchless scatter: invalid elements go to dump slot 48.
    const int jg = 16 * blk + Nl;
    const bool rowok = (krow >= 0);
#pragma unroll
    for (int vv = 0; vv < 8; ++vv) {
      const int M = vv + 8 * hi;
      const int d = M + 32 - jg;
      const bool valid = rowok && ((unsigned)d <= 32u);
      const int slot = valid ? d : 48;
      scw[w][M][slot] = c[vv];
    }
  }

  // ---- Sparse offsets: need only diag of Q_tile * K[t-d..t-d+15]^T.
#pragma unroll
  for (int si = 0; si < 11; ++si) {
    const int d    = OFFS[33 + si];
    const int krow = t - d + Nl;
    v16h b0 = {}, b1 = {};
    if (krow >= 0) {
      const float* kr = kb + (size_t)krow * HDIM;
      cvt8(kr +      16 * hi,     b0, 0);
      cvt8(kr +      16 * hi + 8, b0, 8);
      cvt8(kr + 32 + 16 * hi,     b1, 0);
      cvt8(kr + 32 + 16 * hi + 8, b1, 8);
      __builtin_prefetch(vb + (size_t)krow * HDIM, 0, 1);
    }
    v8f c = {};
    c = __builtin_amdgcn_wmma_f32_16x16x32_f16(false, a0, false, b0, (short)0, c, false, false);
    c = __builtin_amdgcn_wmma_f32_16x16x32_f16(false, a1, false, b1, (short)0, c, false, false);
    // Diagonal element (M==N==Nl) lives in lane with l>>4 == Nl>>3, VGPR Nl&7.
    // Branchless: cndmask-select c[Nl&7], single store to target or dump slot.
    float dval = c[0];
#pragma unroll
    for (int vv = 1; vv < 8; ++vv) dval = ((Nl & 7) == vv) ? c[vv] : dval;
    const bool valid = (krow >= 0) && ((Nl >> 3) == hi);
    const int slot = valid ? (33 + si) : 48;
    scw[w][Nl][slot] = dval;
  }

  __syncthreads();

  // ---- Phase 2: softmax over 44 offsets + weighted v-gather. 2 lanes per query.
  const int r    = lane >> 1;
  const int half = lane & 1;          // 32 output columns per lane
  const int n    = t + r;
  const float* srow = scw[w][r];

  float m = -__builtin_inff();
  for (int i = 0; i < NOFF; ++i) {
    float s = srow[i] + bias[i * NHEAD + hh];   // -inf + bias stays -inf
    m = fmaxf(m, s);
  }

  float acc[32];
#pragma unroll
  for (int j = 0; j < 32; ++j) acc[j] = 0.f;

  float den = 0.f;
  for (int i = 0; i < NOFF; ++i) {
    float s = srow[i] + bias[i * NHEAD + hh];
    float p = __expf(s - m);          // exp(-inf) = 0 for masked offsets
    den += p;
    const int row = n - OFFS[i];
    if (row >= 0) {
      const v4f* vp = (const v4f*)(vb + (size_t)row * HDIM + 32 * half);
#pragma unroll
      for (int j = 0; j < 8; ++j) {
        v4f x = vp[j];
#pragma unroll
        for (int e = 0; e < 4; ++e) acc[4 * j + e] = fmaf(p, x[e], acc[4 * j + e]);
      }
    }
  }

  const float inv = 1.f / den;
  v4f* ov = (v4f*)(out + base + (size_t)n * HDIM + 32 * half);
#pragma unroll
  for (int j = 0; j < 8; ++j) {
    v4f x;
#pragma unroll
    for (int e = 0; e < 4; ++e) x[e] = acc[4 * j + e] * inv;
    ov[j] = x;
  }
}

extern "C" void kernel_launch(void* const* d_in, const int* in_sizes, int n_in,
                              void* d_out, int out_size, void* d_ws, size_t ws_size,
                              hipStream_t stream) {
  (void)n_in; (void)out_size; (void)d_ws; (void)ws_size;
  const float* q    = (const float*)d_in[0];
  const float* k    = (const float*)d_in[1];
  const float* v    = (const float*)d_in[2];
  const float* bias = (const float*)d_in[3];
  float* out = (float*)d_out;

  const int BH = in_sizes[0] / (NQ * HDIM);   // B*H = 64
  const int blocks = BH * (NQ / 128);         // 128 queries per 256-thread block
  dsqg_attn_wmma_kernel<<<dim3(blocks), dim3(256), 0, stream>>>(q, k, v, bias, out);
}
